// Decoder_70334384439561
// MI455X (gfx1250) — matrix operand order repaired and checked by reading, not compiled, not run on hardware
//
#include <hip/hip_runtime.h>
#include <hip/hip_bf16.h>
#include <stdint.h>

// ---------------------------------------------------------------------------
// Tacotron2-style decoder for MI455X (gfx1250, wave32).
//  - bf16 WMMA (v_wmma_f32_16x16x32_bf16) for both LSTM gate GEMMs
//  - persistent kernel (64 WGs) with software grid barrier: the 500-step scan
//    is strictly sequential, so launch latency must be amortized to ~zero.
//  - all weights packed once to bf16 WMMA-fragment layout (L2-resident,
//    halves per-step L2 traffic vs f32).
//  - GEMM k-loop split into compile-time-constant region loops: inner loop is
//    just b128 loads + pointer bumps + 2 WMMAs (no per-iteration branching).
// ---------------------------------------------------------------------------

typedef __attribute__((ext_vector_type(16))) __bf16 v16bf;
typedef __attribute__((ext_vector_type(8)))  float  v8f;
typedef __attribute__((ext_vector_type(4)))  unsigned int u32x4;

#define NWG   64
#define NTHR  256
#define B_    64
#define TENC  512
#define TDEC  500
#define ENCD  512
#define ATTD  1024
#define DECD  1024
#define PRED  256
#define ADIM  128
#define MELR  160
#define KT1   56      // (256 + 512 + 1024) / 32
#define KT2   80      // (1024 + 512 + 1024) / 32
#define NTILES 256    // 4096 / 16

union FragBF { v16bf v; u32x4 q[2]; };
union AccF   { v8f v; float f[8]; };

__device__ __forceinline__ unsigned short f2bf(float x) {
  unsigned u = __float_as_uint(x);
  unsigned r = u + 0x7fffu + ((u >> 16) & 1u);   // round-to-nearest-even
  return (unsigned short)(r >> 16);
}
__device__ __forceinline__ float sigmoidf(float x) {
  return 1.0f / (1.0f + expf(-x));
}

// ------------------------- software grid barrier ---------------------------
__device__ __forceinline__ void grid_sync(unsigned* bar, unsigned* gen) {
  __syncthreads();
  __threadfence();
  *gen += 1;
  if (threadIdx.x == 0) {
    atomicAdd(bar, 1u);
    const unsigned target = (*gen) * (unsigned)NWG;
    while (atomicAdd(bar, 0u) < target) {
      __builtin_amdgcn_s_sleep(1);
    }
  }
  __syncthreads();
  __threadfence();
}

// ------------------------- WMMA gate GEMM ----------------------------------
// One A-source region with a compile-time trip count: straight-line inner
// loop -> b128 loads, pointer bumps, prefetch, 2 WMMAs.
template<int KTN>
__device__ __forceinline__ void gemm_region(
    const unsigned short*& pb,
    const unsigned short* pa0, const unsigned short* pa1,
    AccF& acc0, AccF& acc1)
{
#pragma unroll 2
  for (int kt = 0; kt < KTN; ++kt) {
    FragBF a0f, a1f, bfr;
    a0f.q[0] = *(const u32x4*)(pa0);        // K = k0 .. k0+7
    a0f.q[1] = *(const u32x4*)(pa0 + 16);   // K = k0+16 .. k0+23
    a1f.q[0] = *(const u32x4*)(pa1);
    a1f.q[1] = *(const u32x4*)(pa1 + 16);
    bfr.q[0] = *(const u32x4*)(pb);
    bfr.q[1] = *(const u32x4*)(pb + 8);
    __builtin_prefetch(pb + 512, 0, 1);     // next k-tile B fragment
    pb += 512; pa0 += 32; pa1 += 32;

    acc0.v = __builtin_amdgcn_wmma_f32_16x16x32_bf16(
        false, a0f.v, false, bfr.v, (short)0, acc0.v, false, false);
    acc1.v = __builtin_amdgcn_wmma_f32_16x16x32_bf16(
        false, a1f.v, false, bfr.v, (short)0, acc1.v, false, false);
  }
}

// gates[64,4096] = A[64,K] (bf16, three row-major regions) x packed bf16 B
// (WMMA fragment-major). WG owns 4 ntiles (64 cols); each of the 8 waves
// computes 2 row-tiles x 1 col-tile.
template<int N0, int N1, int N2>
__device__ __forceinline__ void gemm_gates3(
    int tid, int wg,
    const unsigned short* __restrict__ wB,
    const unsigned short* __restrict__ A0, int s0,
    const unsigned short* __restrict__ A1, int s1,
    const unsigned short* __restrict__ A2, int s2,
    float* __restrict__ gates)
{
  constexpr int KT = N0 + N1 + N2;
  const int lane  = tid & 31;
  const int wave  = tid >> 5;
  const int ntile = wg * 4 + (wave & 3);
  const int mt0   = (wave >> 2) * 2;
  const int half  = lane >> 4;
  const int mrow0 = mt0 * 16 + (lane & 15);
  const int mrow1 = mrow0 + 16;
  const int aoff  = half * 8;

  AccF acc0, acc1;
#pragma unroll
  for (int v = 0; v < 8; ++v) { acc0.f[v] = 0.0f; acc1.f[v] = 0.0f; }

  const unsigned short* pb = wB + ((size_t)ntile * KT) * 512 + (size_t)lane * 16;

  gemm_region<N0>(pb, A0 + (size_t)mrow0 * s0 + aoff,
                      A0 + (size_t)mrow1 * s0 + aoff, acc0, acc1);
  gemm_region<N1>(pb, A1 + (size_t)mrow0 * s1 + aoff,
                      A1 + (size_t)mrow1 * s1 + aoff, acc0, acc1);
  gemm_region<N2>(pb, A2 + (size_t)mrow0 * s2 + aoff,
                      A2 + (size_t)mrow1 * s2 + aoff, acc0, acc1);

  const int col = ntile * 16 + (lane & 15);
  const int rb0 = mt0 * 16 + half * 8;
#pragma unroll
  for (int v = 0; v < 8; ++v) {
    gates[(size_t)(rb0 + v) * 4096 + col]      = acc0.f[v];
    gates[(size_t)(rb0 + 16 + v) * 4096 + col] = acc1.f[v];
  }
}

// ------------------------- LSTM pointwise ----------------------------------
__device__ __forceinline__ void lstm_pointwise(
    int gid, const float* __restrict__ gates, const float* __restrict__ bias,
    float* __restrict__ c, float* __restrict__ h, unsigned short* __restrict__ hbf)
{
  for (int e = gid; e < B_ * 1024; e += NWG * NTHR) {
    const int b = e >> 10, j = e & 1023;
    const float* g = gates + (size_t)b * 4096;
    const float gi = g[j]        + bias[j];
    const float gf = g[1024 + j] + bias[1024 + j];
    const float gg = g[2048 + j] + bias[2048 + j];
    const float go = g[3072 + j] + bias[3072 + j];
    const float c2 = sigmoidf(gf) * c[e] + sigmoidf(gi) * tanhf(gg);
    const float hh = sigmoidf(go) * tanhf(c2);
    c[e] = c2; h[e] = hh; hbf[e] = f2bf(hh);
  }
}

// ------------------------- persistent decoder kernel -----------------------
struct DecP {
  const float* inputs; const unsigned char* mask;
  const float* wq; const float* wv; const float* bv;
  const float* loc_conv; const float* loc_dense;
  const float* proj_w; const float* proj_b; const float* stop_w; const float* stop_b;
  const unsigned short* wB1; const unsigned short* wB2;
  const float* bias1; const float* bias2;
  const unsigned short* pre_bf; const float* procin;
  float* gates; float* ah; float* ac; float* dh; float* dc;
  unsigned short* ah_bf; unsigned short* dh_bf;
  float* ctx; unsigned short* ctx_bf;
  float* aw; float* awc;
  float* out_mel; float* out_stop; float* out_align;
  unsigned* bar;
};

__global__ __launch_bounds__(NTHR, 1) void k_decoder(DecP P) {
  const int tid = threadIdx.x;
  const int wg  = blockIdx.x;           // 0..63 ; also the batch row this WG owns
  const int gid = wg * NTHR + tid;
  unsigned gen = 0;

  __shared__ float s_ldense[32 * 128];
  __shared__ float s_lconv[32 * 62];
  __shared__ float s_pq[128];
  __shared__ float s_e[512];
  __shared__ float s_aw[542];
  __shared__ float s_awc[542];
  __shared__ float s_red[256];
  __shared__ float s_dec[160];

  // stage attention weights in LDS once (first use is after a grid_sync)
  for (int i = tid; i < 32 * 128; i += NTHR) s_ldense[i] = P.loc_dense[i];
  for (int i = tid; i < 32 * 62;  i += NTHR) s_lconv[i]  = P.loc_conv[i];

  const float bv0 = P.bv[0];

  for (int t = 0; t < TDEC; ++t) {
    // ---- Phase A: attention-LSTM gates:  [pre[t] | ctx | ah] x W1 ----
    const unsigned short* pre_t = P.pre_bf + (size_t)t * B_ * PRED;
    gemm_gates3<8, 16, 32>(tid, wg, P.wB1,
                           pre_t, PRED,
                           P.ctx_bf, ENCD,
                           P.ah_bf, ATTD,
                           P.gates);
    grid_sync(P.bar, &gen);

    // ---- Phase B: attention-LSTM pointwise -> ah, ac, ah_bf ----
    lstm_pointwise(gid, P.gates, P.bias1, P.ac, P.ah, P.ah_bf);
    grid_sync(P.bar, &gen);

    // ---- Phase C: location-sensitive attention (WG wg owns row b=wg) ----
    {
      const int b = wg;
      for (int i = tid; i < 512; i += NTHR) {
        s_aw[i + 15]  = P.aw[b * 512 + i];
        s_awc[i + 15] = P.awc[b * 512 + i];
      }
      if (tid < 15) { s_aw[tid] = 0.f; s_awc[tid] = 0.f;
                      s_aw[527 + tid] = 0.f; s_awc[527 + tid] = 0.f; }
      // pq = ah[b] @ wq   (128 outputs)
      if (tid < 128) {
        float s = 0.f;
        const float* ahr = P.ah + (size_t)b * ATTD;
#pragma unroll 8
        for (int k = 0; k < ATTD; ++k) s += ahr[k] * P.wq[(size_t)k * ADIM + tid];
        s_pq[tid] = s;
      }
      __syncthreads();

      // energies
      const unsigned char* mrow = P.mask + (size_t)b * TENC;
      for (int t0 = tid; t0 < TENC; t0 += NTHR) {
        float loc[32];
#pragma unroll
        for (int f = 0; f < 32; ++f) {
          float a = 0.f;
          const float* cw = s_lconv + f * 62;
#pragma unroll
          for (int k = 0; k < 31; ++k)
            a += s_aw[t0 + k] * cw[k] + s_awc[t0 + k] * cw[31 + k];
          loc[f] = a;
        }
        float ev = 0.f;
        const float* pin = P.procin + ((size_t)b * TENC + t0) * ADIM;
        for (int d = 0; d < ADIM; ++d) {
          float s = s_pq[d] + pin[d];
#pragma unroll
          for (int f = 0; f < 32; ++f) s += loc[f] * s_ldense[f * ADIM + d];
          ev += tanhf(s) * P.wv[d];
        }
        ev += bv0;
        if (!mrow[t0]) ev = -1e9f;
        s_e[t0] = ev;
      }
      __syncthreads();

      // softmax over 512
      const float v0 = s_e[tid], v1 = s_e[tid + 256];
      s_red[tid] = fmaxf(v0, v1);
      for (int o = 128; o > 0; o >>= 1) {
        __syncthreads();
        if (tid < o) s_red[tid] = fmaxf(s_red[tid], s_red[tid + o]);
      }
      __syncthreads();
      const float mx = s_red[0];
      __syncthreads();
      const float p0 = expf(v0 - mx), p1 = expf(v1 - mx);
      s_red[tid] = p0 + p1;
      for (int o = 128; o > 0; o >>= 1) {
        __syncthreads();
        if (tid < o) s_red[tid] += s_red[tid + o];
      }
      __syncthreads();
      const float inv = 1.0f / s_red[0];
      __syncthreads();
      s_e[tid] = p0 * inv; s_e[tid + 256] = p1 * inv;
      __syncthreads();

      for (int i = tid; i < 512; i += NTHR) {
        const float a2 = s_e[i];
        P.aw[b * 512 + i]  = a2;
        P.awc[b * 512 + i] = s_awc[i + 15] + a2;
        P.out_align[((size_t)b * TDEC + t) * TENC + i] = a2;
      }
      // context: ctx[b,d] = sum_t aw2[t] * inputs[b,t,d]
      for (int d = tid; d < ENCD; d += NTHR) {
        float s = 0.f;
        const float* inp = P.inputs + (size_t)b * TENC * ENCD + d;
        for (int tt = 0; tt < TENC; ++tt) s += s_e[tt] * inp[(size_t)tt * ENCD];
        P.ctx[b * ENCD + d]    = s;
        P.ctx_bf[b * ENCD + d] = f2bf(s);
      }
    }
    grid_sync(P.bar, &gen);

    // ---- Phase D: decoder-LSTM gates:  [ah | ctx | dh] x W2 ----
    gemm_gates3<32, 16, 32>(tid, wg, P.wB2,
                            P.ah_bf, ATTD,
                            P.ctx_bf, ENCD,
                            P.dh_bf, DECD,
                            P.gates);
    grid_sync(P.bar, &gen);

    // ---- Phase E: decoder-LSTM pointwise -> dh, dc, dh_bf ----
    lstm_pointwise(gid, P.gates, P.bias2, P.dc, P.dh, P.dh_bf);
    grid_sync(P.bar, &gen);

    // ---- Phase F: projection + stop token (WG wg owns row b=wg) ----
    {
      const int b = wg;
      if (tid < MELR) {
        float s = P.proj_b[tid];
        const float* dhr = P.dh + (size_t)b * DECD;
#pragma unroll 8
        for (int k = 0; k < DECD; ++k) s += dhr[k] * P.proj_w[(size_t)k * MELR + tid];
        const float* cxr = P.ctx + (size_t)b * ENCD;
#pragma unroll 8
        for (int k = 0; k < ENCD; ++k) s += cxr[k] * P.proj_w[(size_t)(DECD + k) * MELR + tid];
        s_dec[tid] = s;
        const int r = tid / 80, m = tid % 80;
        P.out_mel[(size_t)b * 80000 + (size_t)m * 1000 + (size_t)t * 2 + r] = s;
      }
      __syncthreads();
      float p = 0.f;
      for (int idx = tid; idx < DECD + MELR; idx += NTHR)
        p += (idx < DECD ? P.dh[(size_t)b * DECD + idx] : s_dec[idx - DECD]) * P.stop_w[idx];
      s_red[tid] = p;
      for (int o = 128; o > 0; o >>= 1) {
        __syncthreads();
        if (tid < o) s_red[tid] += s_red[tid + o];
      }
      __syncthreads();
      if (tid == 0) P.out_stop[(size_t)b * TDEC + t] = s_red[0] + P.stop_b[0];
      __syncthreads();
    }
    // no barrier needed here: next step's first conflicting write (ctx in
    // phase C) is separated from this phase by two grid barriers.
  }
}

// ------------------------- one-time prep kernels ---------------------------

// Pack stacked [W1 ; W2] (row-major [4096, k1]/[4096, k2], gates = x @ W^T)
// into bf16 WMMA B-fragment layout: frag(ntile,kt) -> 32 lanes x 16 bf16.
// lane l holds column n = ntile*16 + (l&15); element e holds
// k = kt*32 + (l>>4)*8 + e for e < 8, kt*32 + 16 + (l>>4)*8 + (e-8) for e >= 8.
__global__ void k_pack_w(const float* __restrict__ w1, int k1,
                         const float* __restrict__ w2, int k2,
                         int KT, unsigned short* __restrict__ out) {
  const size_t idx = (size_t)blockIdx.x * blockDim.x + threadIdx.x;
  const size_t total = (size_t)NTILES * KT * 512;
  if (idx >= total) return;
  const int e = (int)(idx & 15);
  const int l = (int)((idx >> 4) & 31);
  const size_t f = idx >> 9;
  const int kt = (int)(f % KT);
  const int nt = (int)(f / KT);
  const int k = kt * 32 + (l >> 4) * 8 + (e < 8 ? e : 8 + e);
  const int n = nt * 16 + (l & 15);
  const float v = (k < k1) ? w1[(size_t)n * k1 + k]
                           : w2[(size_t)n * k2 + (k - k1)];
  out[idx] = f2bf(v);
}

__global__ void k_bias(const float* a_bih, const float* a_bhh,
                       const float* d_bih, const float* d_bhh,
                       float* bias1, float* bias2) {
  const int i = blockIdx.x * blockDim.x + threadIdx.x;
  if (i < 4096) { bias1[i] = a_bih[i] + a_bhh[i]; bias2[i] = d_bih[i] + d_bhh[i]; }
}

// prenet layer 1: h1[t,b,p] = relu(mem[t,b,:160] @ pw1[:,p]); mem[0]=go frame
__global__ void k_prenet1(const float* __restrict__ memories,
                          const float* __restrict__ go,
                          const float* __restrict__ pw1,
                          float* __restrict__ h1) {
  const size_t idx = (size_t)blockIdx.x * blockDim.x + threadIdx.x;
  if (idx >= (size_t)TDEC * B_ * PRED) return;
  const int p = (int)(idx & 255);
  const size_t tb = idx >> 8;
  const int b = (int)(tb & 63);
  const int t = (int)(tb >> 6);
  const float* src = (t == 0) ? go
                              : (memories + (size_t)b * 80000 + (size_t)(t - 1) * MELR);
  float s = 0.f;
#pragma unroll 4
  for (int j = 0; j < MELR; ++j) s += src[j] * pw1[(size_t)j * PRED + p];
  h1[idx] = fmaxf(s, 0.f);
}

// prenet layer 2 -> bf16
__global__ void k_prenet2(const float* __restrict__ h1,
                          const float* __restrict__ pw2,
                          unsigned short* __restrict__ pre_bf) {
  const size_t idx = (size_t)blockIdx.x * blockDim.x + threadIdx.x;
  if (idx >= (size_t)TDEC * B_ * PRED) return;
  const int q = (int)(idx & 255);
  const size_t tb = idx >> 8;
  const float* src = h1 + tb * PRED;
  float s = 0.f;
#pragma unroll 4
  for (int p = 0; p < PRED; ++p) s += src[p] * pw2[(size_t)p * PRED + q];
  pre_bf[idx] = f2bf(fmaxf(s, 0.f));
}

// proc_in[b,t,d] = inputs[b,t,:] @ winp[:,d]
__global__ void k_procin(const float* __restrict__ inputs,
                         const float* __restrict__ winp,
                         float* __restrict__ procin) {
  const size_t idx = (size_t)blockIdx.x * blockDim.x + threadIdx.x;
  if (idx >= (size_t)B_ * TENC * ADIM) return;
  const int d = (int)(idx & 127);
  const size_t bt = idx >> 7;
  const float* src = inputs + bt * ENCD;
  float s = 0.f;
#pragma unroll 4
  for (int k = 0; k < ENCD; ++k) s += src[k] * winp[(size_t)k * ADIM + d];
  procin[idx] = s;
}

__global__ void k_init(const float* att_init, const float* dec_init,
                       float* ah, float* ac, float* dh, float* dc,
                       unsigned short* ah_bf, unsigned short* dh_bf,
                       float* ctx, unsigned short* ctx_bf,
                       float* aw, float* awc) {
  const int idx = blockIdx.x * blockDim.x + threadIdx.x;
  if (idx < B_ * 1024) {
    const int j = idx & 1023;
    const float a = att_init[j], d = dec_init[j];
    ah[idx] = a; ah_bf[idx] = f2bf(a); ac[idx] = 0.f;
    dh[idx] = d; dh_bf[idx] = f2bf(d); dc[idx] = 0.f;
  }
  if (idx < B_ * ENCD) {
    ctx[idx] = 0.f; ctx_bf[idx] = 0;
    aw[idx] = 0.f;  awc[idx] = 0.f;
  }
}

// ---------------------------------------------------------------------------
extern "C" void kernel_launch(void* const* d_in, const int* in_sizes, int n_in,
                              void* d_out, int out_size, void* d_ws, size_t ws_size,
                              hipStream_t stream) {
  (void)in_sizes; (void)n_in; (void)out_size; (void)ws_size;

  const float* inputs   = (const float*)d_in[0];
  const float* memories = (const float*)d_in[1];
  const unsigned char* mask = (const unsigned char*)d_in[2];
  const float* pw1      = (const float*)d_in[3];
  const float* pw2      = (const float*)d_in[4];
  const float* arnn_wih = (const float*)d_in[5];
  const float* arnn_whh = (const float*)d_in[6];
  const float* arnn_bih = (const float*)d_in[7];
  const float* arnn_bhh = (const float*)d_in[8];
  const float* drnn_wih = (const float*)d_in[9];
  const float* drnn_whh = (const float*)d_in[10];
  const float* drnn_bih = (const float*)d_in[11];
  const float* drnn_bhh = (const float*)d_in[12];
  const float* wq       = (const float*)d_in[13];
  const float* winp     = (const float*)d_in[14];
  const float* wv       = (const float*)d_in[15];
  const float* bv       = (const float*)d_in[16];
  const float* loc_conv  = (const float*)d_in[17];
  const float* loc_dense = (const float*)d_in[18];
  const float* proj_w   = (const float*)d_in[19];
  const float* proj_b   = (const float*)d_in[20];
  const float* stop_w   = (const float*)d_in[21];
  const float* stop_b   = (const float*)d_in[22];
  const float* go_frame = (const float*)d_in[23];
  const float* att_init = (const float*)d_in[24];
  const float* dec_init = (const float*)d_in[25];

  char* ws = (char*)d_ws;
  size_t off = 0;
  auto carve = [&](size_t bytes) -> void* {
    void* p = ws + off;
    off += (bytes + 255) & ~(size_t)255;
    return p;
  };

  unsigned* bar        = (unsigned*)carve(256);
  unsigned short* wB1  = (unsigned short*)carve((size_t)NTILES * KT1 * 512 * 2);
  unsigned short* wB2  = (unsigned short*)carve((size_t)NTILES * KT2 * 512 * 2);
  float* bias1         = (float*)carve(4096 * 4);
  float* bias2         = (float*)carve(4096 * 4);
  unsigned short* pre_bf = (unsigned short*)carve((size_t)TDEC * B_ * PRED * 2);
  float* procin        = (float*)carve((size_t)B_ * TENC * ADIM * 4);
  float* h1            = (float*)carve((size_t)TDEC * B_ * PRED * 4);
  float* gates         = (float*)carve((size_t)B_ * 4096 * 4);
  float* ah            = (float*)carve((size_t)B_ * 1024 * 4);
  float* ac            = (float*)carve((size_t)B_ * 1024 * 4);
  float* dh            = (float*)carve((size_t)B_ * 1024 * 4);
  float* dc            = (float*)carve((size_t)B_ * 1024 * 4);
  unsigned short* ah_bf = (unsigned short*)carve((size_t)B_ * 1024 * 2);
  unsigned short* dh_bf = (unsigned short*)carve((size_t)B_ * 1024 * 2);
  float* ctx           = (float*)carve((size_t)B_ * ENCD * 4);
  unsigned short* ctx_bf = (unsigned short*)carve((size_t)B_ * ENCD * 2);
  float* aw            = (float*)carve((size_t)B_ * TENC * 4);
  float* awc           = (float*)carve((size_t)B_ * TENC * 4);

  float* out_mel   = (float*)d_out;                    // [64, 80, 1000]
  float* out_stop  = out_mel + (size_t)B_ * 80 * 1000; // [64, 500]
  float* out_align = out_stop + (size_t)B_ * TDEC;     // [64, 500, 512]

  hipMemsetAsync(bar, 0, 256, stream);

  {
    const size_t tot1 = (size_t)NTILES * KT1 * 512;
    k_pack_w<<<(unsigned)((tot1 + NTHR - 1) / NTHR), NTHR, 0, stream>>>(
        arnn_wih, PRED + ENCD, arnn_whh, ATTD, KT1, wB1);
    const size_t tot2 = (size_t)NTILES * KT2 * 512;
    k_pack_w<<<(unsigned)((tot2 + NTHR - 1) / NTHR), NTHR, 0, stream>>>(
        drnn_wih, ATTD + ENCD, drnn_whh, DECD, KT2, wB2);
  }
  k_bias<<<16, NTHR, 0, stream>>>(arnn_bih, arnn_bhh, drnn_bih, drnn_bhh,
                                  bias1, bias2);
  {
    const size_t tot = (size_t)TDEC * B_ * PRED;
    const unsigned g = (unsigned)((tot + NTHR - 1) / NTHR);
    k_prenet1<<<g, NTHR, 0, stream>>>(memories, go_frame, pw1, h1);
    k_prenet2<<<g, NTHR, 0, stream>>>(h1, pw2, pre_bf);
  }
  {
    const size_t tot = (size_t)B_ * TENC * ADIM;
    k_procin<<<(unsigned)((tot + NTHR - 1) / NTHR), NTHR, 0, stream>>>(
        inputs, winp, procin);
  }
  k_init<<<(B_ * 1024 + NTHR - 1) / NTHR, NTHR, 0, stream>>>(
      att_init, dec_init, ah, ac, dh, dc, ah_bf, dh_bf, ctx, ctx_bf, aw, awc);

  DecP P;
  P.inputs = inputs; P.mask = mask;
  P.wq = wq; P.wv = wv; P.bv = bv;
  P.loc_conv = loc_conv; P.loc_dense = loc_dense;
  P.proj_w = proj_w; P.proj_b = proj_b; P.stop_w = stop_w; P.stop_b = stop_b;
  P.wB1 = wB1; P.wB2 = wB2; P.bias1 = bias1; P.bias2 = bias2;
  P.pre_bf = pre_bf; P.procin = procin;
  P.gates = gates; P.ah = ah; P.ac = ac; P.dh = dh; P.dc = dc;
  P.ah_bf = ah_bf; P.dh_bf = dh_bf;
  P.ctx = ctx; P.ctx_bf = ctx_bf; P.aw = aw; P.awc = awc;
  P.out_mel = out_mel; P.out_stop = out_stop; P.out_align = out_align;
  P.bar = bar;

  k_decoder<<<NWG, NTHR, 0, stream>>>(P);
}
